// ObjAfterSeg_22660247454223
// MI455X (gfx1250) — compile-verified, matrix operand order checked by reading, not run on hardware
//
#include <hip/hip_runtime.h>

typedef __attribute__((ext_vector_type(16))) _Float16 v16h;
typedef __attribute__((ext_vector_type(8)))  _Float16 v8h;
typedef __attribute__((ext_vector_type(8)))  float    v8f;

union Frag16 { v16h v; v8h h[2]; };

// ---------------------------------------------------------------------------
// zero-fill (dwords) -- clears padded activation region each launch
// ---------------------------------------------------------------------------
__global__ __launch_bounds__(256)
void fill_zero_kernel(unsigned int* __restrict__ p, int n) {
    int i = blockIdx.x * blockDim.x + threadIdx.x;
    if (i < n) p[i] = 0u;
}

// ---------------------------------------------------------------------------
// input: NCHW f32 [2,3,128,128] -> NHWC padded f16 [2][129][129][3]
// ---------------------------------------------------------------------------
__global__ __launch_bounds__(256)
void pack_input_kernel(const float* __restrict__ x, _Float16* __restrict__ xh) {
    int i = blockIdx.x * blockDim.x + threadIdx.x;
    if (i >= 2 * 128 * 128 * 3) return;
    int ci = i % 3;  int t = i / 3;
    int xc = t % 128; t /= 128;
    int y  = t % 128;
    int b  = t / 128;
    xh[(((size_t)b * 129 + y) * 129 + xc) * 3 + ci] =
        (_Float16)x[(((size_t)b * 3 + ci) * 128 + y) * 128 + xc];
}

// ---------------------------------------------------------------------------
// weights: OIHW f32 [Co,Ci,3,3] -> f16 [Co][Kp], K-order = (kh*3+kw)*Ci + ci,
// zero padded to Kp
// ---------------------------------------------------------------------------
__global__ __launch_bounds__(256)
void pack_wgt_kernel(const float* __restrict__ w, _Float16* __restrict__ wh,
                     int Ci, int Co, int K, int Kp) {
    int i = blockIdx.x * blockDim.x + threadIdx.x;
    if (i >= Co * Kp) return;
    int n  = i / Kp;
    int kp = i - n * Kp;
    _Float16 v = (_Float16)0.0f;
    if (kp < K) {
        int t  = kp / Ci;        // tap = kh*3+kw
        int ci = kp - t * Ci;
        v = (_Float16)w[((size_t)(n * Ci + ci)) * 9 + t];
    }
    wh[i] = v;
}

// ---------------------------------------------------------------------------
// Layers 1..4: stride-2 3x3 conv + bias + ReLU, implicit GEMM, templated dims.
// One wave32 per 16x16 output tile.  K-loop factored as taps(3x3, unrolled)
// x ci-chunks (unroll 2): each chunk = 4x global_load_b128 + 1x v_wmma.
// act  : NHWC padded f16 [2][Hin+1][Win+1][Ci] (border zeroed)
// wgt  : packed f16 [Co][K]
// out  : f16 [2][Ho+1][Wo+1][Co] padded  (OUT_F32=0)
//        f32 [2][Ho*Wo][Co]              (OUT_F32=1)
// ---------------------------------------------------------------------------
template <int Ci, int Hin, int Win, int Co, int Ho, int Wo, int OUT_F32>
__global__ __launch_bounds__(32)
void conv_vec_kernel(const _Float16* __restrict__ act,
                     const _Float16* __restrict__ wgt,
                     const float*    __restrict__ bias,
                     void*           __restrict__ outp)
{
    const int K  = Ci * 9;
    const int hw = Ho * Wo;
    const int W1 = Win + 1;

    const int lane = threadIdx.x & 31;
    const int l16  = lane & 15;
    const int half = lane >> 4;

    // A row for this lane: m -> (b, oy, ox)   (hw, Wo compile-time pow2)
    const int m  = blockIdx.x * 16 + l16;
    const int b  = m / hw;
    const int rm = m % hw;
    const int oy = rm / Wo;
    const int ox = rm % Wo;

    // B col for this lane
    const int n = blockIdx.y * 16 + l16;

    const _Float16* actb = act + ((size_t)b * (Hin + 1) + 2 * oy) * W1 * Ci
                               + (size_t)(2 * ox) * Ci + half * 8;
    const _Float16* wrow = wgt + (size_t)n * K + half * 8;

    v8f acc = {};
    #pragma unroll
    for (int kh = 0; kh < 3; ++kh) {
        #pragma unroll
        for (int kw = 0; kw < 3; ++kw) {
            const _Float16* ap = actb + ((size_t)kh * W1 + kw) * Ci;
            const _Float16* wp = wrow + (size_t)(kh * 3 + kw) * Ci;
            #pragma unroll 2
            for (int c0 = 0; c0 < Ci; c0 += 32) {
                Frag16 af, bf;
                af.h[0] = *(const v8h*)(ap + c0);
                af.h[1] = *(const v8h*)(ap + c0 + 16);
                bf.h[0] = *(const v8h*)(wp + c0);
                bf.h[1] = *(const v8h*)(wp + c0 + 16);
                acc = __builtin_amdgcn_wmma_f32_16x16x32_f16(
                          false, af.v, false, bf.v, (short)0, acc, false, false);
            }
        }
    }

    // Epilogue: C/D layout -> row = half*8 + r, col = l16
    const float bn = bias[n];
    float*    of = (float*)outp;
    _Float16* oh = (_Float16*)outp;
    #pragma unroll
    for (int r = 0; r < 8; ++r) {
        const int mr = blockIdx.x * 16 + half * 8 + r;
        const int bb = mr / hw;
        const int rr = mr % hw;
        const int ey = rr / Wo;
        const int ex = rr % Wo;
        float v = acc[r] + bn;
        v = v > 0.0f ? v : 0.0f;
        if (OUT_F32) {
            of[((size_t)bb * hw + rr) * Co + n] = v;
        } else {
            oh[(((size_t)bb * (Ho + 1) + ey) * (Wo + 1) + ex) * Co + n] = (_Float16)v;
        }
    }
}

// ---------------------------------------------------------------------------
// Layer 0: Ci=3, K=27 (padded to 32).  Single K-chunk; branchless clamped
// gather for A (padded act makes all tap addresses in-bounds), 2x b128 for B.
// ---------------------------------------------------------------------------
__global__ __launch_bounds__(32)
void conv_l0_kernel(const _Float16* __restrict__ act,  // [2][129][129][3]
                    const _Float16* __restrict__ wgt,  // [64][32]
                    const float*    __restrict__ bias, // [64]
                    _Float16*       __restrict__ out)  // [2][65][65][64]
{
    const int Ci = 3, Co = 64, Ho = 64, Wo = 64;
    const int K = 27, Kp = 32;
    const int hw = Ho * Wo;       // 4096
    const int W1 = 129;

    const int lane = threadIdx.x & 31;
    const int l16  = lane & 15;
    const int half = lane >> 4;

    const int m  = blockIdx.x * 16 + l16;
    const int b  = m / hw;
    const int rm = m % hw;
    const int oy = rm / Wo;
    const int ox = rm % Wo;
    const int n  = blockIdx.y * 16 + l16;

    const _Float16* actb = act + ((size_t)b * 129 + 2 * oy) * W1 * Ci + (size_t)(2 * ox) * Ci;
    const _Float16* wrow = wgt + (size_t)n * Kp + half * 8;

    Frag16 af, bf;
    bf.h[0] = *(const v8h*)(wrow);
    bf.h[1] = *(const v8h*)(wrow + 16);

    #pragma unroll
    for (int u = 0; u < 2; ++u) {
        v8h tmp;
        #pragma unroll
        for (int j = 0; j < 8; ++j) {
            const int k   = half * 8 + u * 16 + j;     // 0..31
            const int kc  = (k < K) ? k : 0;           // clamp, select later
            const int tap = kc / Ci;
            const int ci0 = kc - tap * Ci;
            const int kh  = tap / 3;
            const int kw  = tap - 3 * kh;
            const _Float16 v0 = actb[((size_t)kh * W1 + kw) * Ci + ci0];
            tmp[j] = (k < K) ? v0 : (_Float16)0.0f;
        }
        af.h[u] = tmp;
    }

    v8f acc = {};
    acc = __builtin_amdgcn_wmma_f32_16x16x32_f16(
              false, af.v, false, bf.v, (short)0, acc, false, false);

    const float bn = bias[n];
    #pragma unroll
    for (int r = 0; r < 8; ++r) {
        const int mr = blockIdx.x * 16 + half * 8 + r;
        const int bb = mr / hw;
        const int rr = mr % hw;
        const int ey = rr / Wo;
        const int ex = rr % Wo;
        float v = acc[r] + bn;
        v = v > 0.0f ? v : 0.0f;
        out[(((size_t)bb * (Ho + 1) + ey) * (Wo + 1) + ex) * Co + n] = (_Float16)v;
    }
}

// ---------------------------------------------------------------------------
// Heads on the 32 unique low-res cells (nearest-upsample factored out).
// feat: NHWC f32 [2][16][2048] -> 6 dots of length 2048 per cell.
// ---------------------------------------------------------------------------
__global__ __launch_bounds__(256)
void heads_kernel(const float* __restrict__ feat,
                  const float* __restrict__ seg_w, const float* __restrict__ seg_b,
                  const float* __restrict__ sc_w,  const float* __restrict__ sc_b,
                  const float* __restrict__ bb_w,  const float* __restrict__ bb_b,
                  float* __restrict__ head)
{
    __shared__ float red[6][256];
    const int cell = blockIdx.x;            // 0..31  (b*16 + cy*4 + cx)
    const int t    = threadIdx.x;
    const float* fc = feat + (size_t)cell * 2048;

    float a0 = 0.f, a1 = 0.f, a2 = 0.f, a3 = 0.f, a4 = 0.f, a5 = 0.f;
    for (int c = t; c < 2048; c += 256) {
        const float f = fc[c];
        a0 += f * seg_w[c];
        a1 += f * sc_w[c];
        a2 += f * bb_w[c];
        a3 += f * bb_w[2048 + c];
        a4 += f * bb_w[2 * 2048 + c];
        a5 += f * bb_w[3 * 2048 + c];
    }
    red[0][t] = a0; red[1][t] = a1; red[2][t] = a2;
    red[3][t] = a3; red[4][t] = a4; red[5][t] = a5;
    __syncthreads();
    for (int s = 128; s > 0; s >>= 1) {
        if (t < s) {
            #pragma unroll
            for (int j = 0; j < 6; ++j) red[j][t] += red[j][t + s];
        }
        __syncthreads();
    }
    if (t < 6) {
        const float bv = (t == 0) ? seg_b[0] : (t == 1) ? sc_b[0] : bb_b[t - 2];
        head[cell * 6 + t] = red[t][0] + bv;
    }
}

// ---------------------------------------------------------------------------
// Per-pixel expansion: sigmoid/threshold, bbox transform with per-pixel grid
// coords, write d_out = scores[2*HW] | bboxes[8*HW] | seg[2*HW].
// ---------------------------------------------------------------------------
__global__ __launch_bounds__(256)
void expand_kernel(const float* __restrict__ head, float* __restrict__ out)
{
    const int p = blockIdx.x * blockDim.x + threadIdx.x;   // 0..32767
    if (p >= 2 * 128 * 128) return;
    const int b   = p >> 14;
    const int rem = p & 16383;
    const int y   = rem >> 7;
    const int x   = rem & 127;
    const int cell = (b << 4) | ((y >> 5) << 2) | (x >> 5);
    const float* h = head + cell * 6;

    const float seg   = h[0];
    const float score = 1.0f / (1.0f + __expf(-h[1]));
    const bool  has   = score >= 0.5f;
    const float sc    = has ? score : 0.0f;

    const float xc = ((float)x + h[2]) * (1.0f / 128.0f);
    const float yc = ((float)y + h[3]) * (1.0f / 128.0f);
    const float bw = __expf(h[4]);
    const float bh = __expf(h[5]);
    const float b0 = has ? (xc - 0.5f * bw) : 0.0f;
    const float b1 = has ? (yc - 0.5f * bh) : 0.0f;
    const float b2 = has ? (xc + 0.5f * bw) : 0.0f;
    const float b3 = has ? (yc + 0.5f * bh) : 0.0f;

    const int HW = 128 * 128;
    out[b * HW + rem] = sc;                              // scores
    float* bb = out + 2 * HW;                            // bboxes
    bb[(b * 4 + 0) * HW + rem] = b0;
    bb[(b * 4 + 1) * HW + rem] = b1;
    bb[(b * 4 + 2) * HW + rem] = b2;
    bb[(b * 4 + 3) * HW + rem] = b3;
    out[10 * HW + b * HW + rem] = seg;                   // seg
}

// ---------------------------------------------------------------------------
extern "C" void kernel_launch(void* const* d_in, const int* in_sizes, int n_in,
                              void* d_out, int out_size, void* d_ws, size_t ws_size,
                              hipStream_t stream) {
    (void)in_sizes; (void)n_in; (void)out_size; (void)ws_size;

    const float* x        = (const float*)d_in[0];
    const float* ew[5]    = { (const float*)d_in[1], (const float*)d_in[3], (const float*)d_in[5],
                              (const float*)d_in[7], (const float*)d_in[9] };
    const float* eb[5]    = { (const float*)d_in[2], (const float*)d_in[4], (const float*)d_in[6],
                              (const float*)d_in[8], (const float*)d_in[10] };
    const float* seg_w    = (const float*)d_in[11];
    const float* seg_b    = (const float*)d_in[12];
    const float* scores_w = (const float*)d_in[13];
    const float* scores_b = (const float*)d_in[14];
    const float* bboxes_w = (const float*)d_in[15];
    const float* bboxes_b = (const float*)d_in[16];

    // --- workspace carve-out (256B aligned) ---
    char* ws = (char*)d_ws;
    size_t off = 0;
    auto carve = [&](size_t bytes) -> void* {
        void* p = ws + off;
        off = (off + bytes + 255) & ~(size_t)255;
        return p;
    };
    // padded NHWC activations (zero-filled each launch)
    _Float16* xh = (_Float16*)carve((size_t)2 * 129 * 129 * 3   * sizeof(_Float16));
    _Float16* a0 = (_Float16*)carve((size_t)2 * 65  * 65  * 64  * sizeof(_Float16));
    _Float16* a1 = (_Float16*)carve((size_t)2 * 33  * 33  * 128 * sizeof(_Float16));
    _Float16* a2 = (_Float16*)carve((size_t)2 * 17  * 17  * 256 * sizeof(_Float16));
    _Float16* a3 = (_Float16*)carve((size_t)2 * 9   * 9   * 512 * sizeof(_Float16));
    const size_t act_bytes = off;                       // region to zero-fill
    // unpadded buffers
    float* feat = (float*)carve((size_t)2 * 16 * 2048 * sizeof(float));
    float* head = (float*)carve((size_t)32 * 6 * sizeof(float));
    // packed f16 weights: [Co][Kp]
    const int Kp0 = 32, K1 = 576, K2 = 1152, K3 = 2304, K4 = 4608;
    _Float16* wh0 = (_Float16*)carve((size_t)64   * Kp0 * sizeof(_Float16));
    _Float16* wh1 = (_Float16*)carve((size_t)128  * K1  * sizeof(_Float16));
    _Float16* wh2 = (_Float16*)carve((size_t)256  * K2  * sizeof(_Float16));
    _Float16* wh3 = (_Float16*)carve((size_t)512  * K3  * sizeof(_Float16));
    _Float16* wh4 = (_Float16*)carve((size_t)2048 * K4  * sizeof(_Float16));

    // 1) zero the padded activation region (borders must be 0)
    {
        const int nd = (int)(act_bytes / 4);
        fill_zero_kernel<<<(nd + 255) / 256, 256, 0, stream>>>((unsigned int*)d_ws, nd);
    }
    // 2) pack input -> NHWC padded f16
    {
        const int n = 2 * 128 * 128 * 3;
        pack_input_kernel<<<(n + 255) / 256, 256, 0, stream>>>(x, xh);
    }
    // 3) pack weights -> f16 [Co][Kp], K-order (kh,kw,ci)
    auto packw = [&](const float* w, _Float16* wh, int Ci, int Co, int Kp) {
        const int n = Co * Kp;
        pack_wgt_kernel<<<(n + 255) / 256, 256, 0, stream>>>(w, wh, Ci, Co, Ci * 9, Kp);
    };
    packw(ew[0], wh0, 3,   64,   Kp0);
    packw(ew[1], wh1, 64,  128,  K1);
    packw(ew[2], wh2, 128, 256,  K2);
    packw(ew[3], wh3, 256, 512,  K3);
    packw(ew[4], wh4, 512, 2048, K4);

    // 4) conv trunk: grid = (M/16, Co/16), one wave32 per tile
    conv_l0_kernel<<<dim3(512, 4), 32, 0, stream>>>(xh, wh0, eb[0], a0);
    conv_vec_kernel<64,  64, 64, 128,  32, 32, 0><<<dim3(128,   8), 32, 0, stream>>>(a0, wh1, eb[1], a1);
    conv_vec_kernel<128, 32, 32, 256,  16, 16, 0><<<dim3( 32,  16), 32, 0, stream>>>(a1, wh2, eb[2], a2);
    conv_vec_kernel<256, 16, 16, 512,  8,  8,  0><<<dim3(  8,  32), 32, 0, stream>>>(a2, wh3, eb[3], a3);
    conv_vec_kernel<512, 8,  8,  2048, 4,  4,  1><<<dim3(  2, 128), 32, 0, stream>>>(a3, wh4, eb[4], feat);

    // 5) heads on the 32 unique cells
    heads_kernel<<<32, 256, 0, stream>>>(feat, seg_w, seg_b, scores_w, scores_b,
                                         bboxes_w, bboxes_b, head);
    // 6) per-pixel expansion
    expand_kernel<<<128, 256, 0, stream>>>(head, (float*)d_out);
}
